// TransformerBlock_5033701671200
// MI455X (gfx1250) — matrix operand order repaired and checked by reading, not compiled, not run on hardware
//
#include <hip/hip_runtime.h>
#include <hip/hip_bf16.h>
#include <math.h>

typedef __bf16 bf16;
typedef __attribute__((ext_vector_type(16))) __bf16 bf16x16;
typedef __attribute__((ext_vector_type(8)))  float  f32x8;

#define BB  2
#define SS  2048
#define EE  128
#define HH  16
#define DD  128
#define DHH 512
#define MM  (BB * SS)   // 4096 rows

__device__ __forceinline__ bf16 to_bf16(float f) { return (bf16)f; }

// ---------------------------------------------------------------------------
// CDNA5 async global->LDS copy (ASYNCcnt-tracked DMA, no VGPR round trip)
// ROCm 7.2 signature: (int4 __device__* src, int4 __shared__* dst, imm off, imm cpol)
// where int4 is a GCC-style vector: int __attribute__((vector_size(16)))
// ---------------------------------------------------------------------------
typedef int gv4i __attribute__((vector_size(16)));

#if __has_builtin(__builtin_amdgcn_global_load_async_to_lds_b128)
__device__ __forceinline__ void async_ld128(bf16* dst, const bf16* src) {
  __builtin_amdgcn_global_load_async_to_lds_b128(
      (__attribute__((address_space(1))) gv4i*)(uintptr_t)src,
      (__attribute__((address_space(3))) gv4i*)(uintptr_t)dst, 0, 0);
}
#else
__device__ __forceinline__ void async_ld128(bf16* dst, const bf16* src) {
  const unsigned lds_off = (unsigned)(uintptr_t)dst;
  asm volatile("global_load_async_to_lds_b128 %0, %1, off"
               :: "v"(lds_off), "v"(src) : "memory");
}
#endif

#if __has_builtin(__builtin_amdgcn_s_wait_asynccnt)
#define WAIT_ASYNC(n) __builtin_amdgcn_s_wait_asynccnt(n)
#else
#define WAIT_ASYNC(n) asm volatile("s_wait_asynccnt %0" :: "i"(n) : "memory")
#endif

// ---------------------------------------------------------------------------
// WMMA fragment helpers (wave32, v_wmma_f32_16x16x32_bf16)
// ---------------------------------------------------------------------------
union FragBF { bf16x16 v; unsigned int u[8]; };

// A-fragment (16x32 bf16, row M = lane&15): K = (j/4)*16 + (lane>>4)*8 + (j%4)*2 (+pos)
__device__ __forceinline__ bf16x16 load_frag_a(const bf16* base, int ldk, int lane) {
  const int row = lane & 15, half = lane >> 4;
  FragBF f;
#pragma unroll
  for (int j = 0; j < 8; ++j) {
    const int k = ((j >> 2) << 4) + half * 8 + ((j & 3) << 1);
    f.u[j] = *(const unsigned int*)(base + row * ldk + k);
  }
  return f.v;
}

// B-fragment (32x16 bf16) from B^T layout [N][K]: col N = lane&15, K = (j/4)*16 + (lane>>4)*8 + (j%4)*2 (+pos)
__device__ __forceinline__ bf16x16 load_frag_b(const bf16* base, int ldk, int lane) {
  const int col = lane & 15, half = lane >> 4;
  FragBF f;
#pragma unroll
  for (int j = 0; j < 8; ++j) {
    const int k = ((j >> 2) << 4) + half * 8 + ((j & 3) << 1);
    f.u[j] = *(const unsigned int*)(base + col * ldk + k);
  }
  return f.v;
}

__device__ __forceinline__ f32x8 wmma_bf16(bf16x16 a, bf16x16 b, f32x8 c) {
  return __builtin_amdgcn_wmma_f32_16x16x32_bf16(false, a, false, b, (short)0, c, false, false);
}

// ---------------------------------------------------------------------------
// Generic bf16 WMMA GEMM:  C[M,N] = A[M,K] * Bt[N,K]^T  (+ fused epilogues)
// block = 128 threads (4 waves); block tile 64x64; double-buffered async stage.
// ---------------------------------------------------------------------------
#define EP_F32            0
#define EP_BF16           1
#define EP_RES_F32        2
#define EP_BIAS_GELU_BF16 3
#define EP_BIAS_RES_F32   4

#define GLDT 40   // padded LDS stride (bf16 elems); 80 B rows (16B-aligned)

__device__ __forceinline__ void gemm_stage_async(
    bf16* dA, bf16* dB, const bf16* A, const bf16* Bt,
    int m0, int n0, int K, int k0, int tid)
{
#pragma unroll
  for (int i = 0; i < 2; ++i) {               // 4 async instrs / thread / stage
    const int idx = tid + i * 128;            // 0..255
    const int r = idx >> 2, c = (idx & 3) * 8;
    async_ld128(dA + r * GLDT + c, A  + (size_t)(m0 + r) * K + k0 + c);
    async_ld128(dB + r * GLDT + c, Bt + (size_t)(n0 + r) * K + k0 + c);
  }
}

template <int EP>
__global__ __launch_bounds__(128) void gemm_wmma(
    const bf16* __restrict__ A, const bf16* __restrict__ Bt,
    int M, int N, int K,
    const float* __restrict__ bias, const float* __restrict__ resid,
    float* __restrict__ outF, bf16* __restrict__ outH)
{
  __shared__ __align__(16) bf16 sA[2][64 * GLDT];
  __shared__ __align__(16) bf16 sB[2][64 * GLDT];
  const int tid = threadIdx.x;
  const int wave = tid >> 5, lane = tid & 31;
  const int m0 = blockIdx.y * 64, n0 = blockIdx.x * 64;
  f32x8 acc[4] = {};

  const int nsteps = K >> 5;
  gemm_stage_async(sA[0], sB[0], A, Bt, m0, n0, K, 0, tid);

  for (int step = 0; step < nsteps; ++step) {
    const int cur = step & 1;
    if (step + 1 < nsteps) {
      gemm_stage_async(sA[cur ^ 1], sB[cur ^ 1], A, Bt, m0, n0, K, (step + 1) * 32, tid);
      WAIT_ASYNC(4);          // current buffer's 4 ops complete; next 4 in flight
    } else {
      WAIT_ASYNC(0);
    }
    __syncthreads();

    const bf16x16 af = load_frag_a(sA[cur] + wave * 16 * GLDT, GLDT, lane);
#pragma unroll
    for (int n = 0; n < 4; ++n) {
      const bf16x16 bfr = load_frag_b(sB[cur] + n * 16 * GLDT, GLDT, lane);
      acc[n] = wmma_bf16(af, bfr, acc[n]);
    }
    __syncthreads();          // protect buf[cur^1] before next iteration refills it
  }

  const int half = lane >> 4, cin = lane & 15;
#pragma unroll
  for (int n = 0; n < 4; ++n) {
    const int col = n0 + n * 16 + cin;
#pragma unroll
    for (int r8 = 0; r8 < 8; ++r8) {
      const int row = m0 + wave * 16 + r8 + 8 * half;
      float v = acc[n][r8];
      if (EP == EP_BIAS_GELU_BF16 || EP == EP_BIAS_RES_F32) v += bias[col];
      if (EP == EP_BIAS_GELU_BF16) v = 0.5f * v * (1.0f + erff(v * 0.70710678118f));
      if (EP == EP_RES_F32 || EP == EP_BIAS_RES_F32) v += resid[(size_t)row * N + col];
      if (EP == EP_BF16 || EP == EP_BIAS_GELU_BF16) outH[(size_t)row * N + col] = to_bf16(v);
      else outF[(size_t)row * N + col] = v;
    }
  }
}

// ---------------------------------------------------------------------------
// Flash attention (causal). Block = 128 threads (4 waves) handles 64 query
// rows of one (b, h). Double-buffered async K/V^T staging; online softmax.
// q,k: bf16 [B,S,H,D]; vT: bf16 [B,H,D,S]; out: bf16 [B,S,H*D]
// ---------------------------------------------------------------------------
#define FLK 136   // sK row stride (272 B, 16B-aligned)
#define FLV 72    // sV row stride (144 B, 16B-aligned)
#define FLP 72

__device__ __forceinline__ void fa_stage_async(
    bf16* dK, bf16* dV, const bf16* kbase, const bf16* vbase, int tid)
{
#pragma unroll
  for (int i = 0; i < 8; ++i) {               // 64x128 key tile: 1024 b128 chunks
    const int idx = tid + i * 128;
    const int r = idx >> 4, c = (idx & 15) * 8;
    async_ld128(dK + r * FLK + c, kbase + (size_t)r * (HH * DD) + c);
  }
#pragma unroll
  for (int i = 0; i < 8; ++i) {               // 128x64 V^T tile: 1024 b128 chunks
    const int idx = tid + i * 128;
    const int r = idx >> 3, c = (idx & 7) * 8;
    async_ld128(dV + r * FLV + c, vbase + (size_t)r * SS + c);
  }
}

__global__ __launch_bounds__(128) void flash_attn(
    const bf16* __restrict__ qb, const bf16* __restrict__ kb,
    const bf16* __restrict__ vt, bf16* __restrict__ out)
{
  __shared__ __align__(16) bf16 sK[2][64 * FLK];
  __shared__ __align__(16) bf16 sV[2][128 * FLV];
  __shared__ __align__(16) bf16 sP[4][16 * FLP];
  __shared__ float sS[4][16 * 64];
  __shared__ float sM[4][16], sL[4][16], sAl[4][16], sRed[4][16][2];

  const int tid = threadIdx.x, wave = tid >> 5, lane = tid & 31;
  const int half = lane >> 4, cin = lane & 15;
  const int qi = blockIdx.x;
  const int b = blockIdx.y / HH, h = blockIdx.y % HH;
  const float scale = 0.08838834764831845f;   // 1/sqrt(128)

  // Stage the 64x128 query tile (sync) into sK[0], pull per-wave A-fragments.
  const bf16* qbase = qb + ((size_t)(b * SS + qi * 64) * HH + h) * DD;
  for (int idx = tid; idx < 64 * 32; idx += 128) {
    const int r = idx >> 5, c = (idx & 31) * 4;
    *(uint2*)(sK[0] + r * FLK + c) = *(const uint2*)(qbase + (size_t)r * (HH * DD) + c);
  }
  if (lane < 16) { sM[wave][lane] = -3.0e38f; sL[wave][lane] = 0.0f; }
  __syncthreads();

  bf16x16 qf[4];
#pragma unroll
  for (int ks = 0; ks < 4; ++ks)
    qf[ks] = load_frag_a(sK[0] + wave * 16 * FLK + ks * 32, FLK, lane);
  f32x8 O[8] = {};
  __syncthreads();

  const bf16* kb_bh = kb + ((size_t)b * SS * HH + h) * DD;        // + s*H*D
  const bf16* vt_bh = vt + ((size_t)(b * HH + h) * DD) * SS;      // + d*S + s

  fa_stage_async(sK[0], sV[0], kb_bh, vt_bh, tid);                // prefetch block 0

  for (int kbk = 0; kbk <= qi; ++kbk) {
    const int cur = kbk & 1;
    if (kbk < qi) {
      fa_stage_async(sK[cur ^ 1], sV[cur ^ 1],
                     kb_bh + (size_t)(kbk + 1) * 64 * (HH * DD),
                     vt_bh + (size_t)(kbk + 1) * 64, tid);
      WAIT_ASYNC(16);         // current tile's 16 ops done; next 16 in flight
    } else {
      WAIT_ASYNC(0);
    }
    __syncthreads();
    const bf16* cK = sK[cur];
    const bf16* cV = sV[cur];

    // Scores: S = q k^T * scale  (16 WMMAs per wave per 64-key block)
#pragma unroll
    for (int n = 0; n < 4; ++n) {
      f32x8 a = {};
#pragma unroll
      for (int ks = 0; ks < 4; ++ks) {
        const bf16x16 kf = load_frag_b(cK + n * 16 * FLK + ks * 32, FLK, lane);
        a = wmma_bf16(qf[ks], kf, a);
      }
#pragma unroll
      for (int r8 = 0; r8 < 8; ++r8) {
        float s = a[r8] * scale;
        if (kbk == qi) {
          const int lcol = n * 16 + cin, lrow = wave * 16 + r8 + 8 * half;
          if (lcol > lrow) s = -1.0e30f;
        }
        sS[wave][(r8 + 8 * half) * 64 + n * 16 + cin] = s;
      }
    }

    // Online softmax (per wave: 16 rows, 2 lanes/row, 32 cols each).
    {
      const int row = lane >> 1, sub = lane & 1;
      float* srow = &sS[wave][row * 64 + sub * 32];
      float bm = -3.0e38f;
#pragma unroll
      for (int c = 0; c < 32; ++c) bm = fmaxf(bm, srow[c]);
      sRed[wave][row][sub] = bm;
      const float bmax = fmaxf(sRed[wave][row][0], sRed[wave][row][1]);
      const float mold = sM[wave][row];
      const float mnew = fmaxf(mold, bmax);
      const float alpha = __expf(mold - mnew);
      float sum = 0.0f;
      bf16* prow = &sP[wave][row * FLP + sub * 32];
#pragma unroll
      for (int c = 0; c < 32; ++c) {
        const float p = __expf(srow[c] - mnew);
        prow[c] = to_bf16(p);
        sum += p;
      }
      sRed[wave][row][sub] = sum;
      if (sub == 0) {
        sM[wave][row]  = mnew;
        sAl[wave][row] = alpha;
        sL[wave][row]  = alpha * sL[wave][row] + sRed[wave][row][0] + sRed[wave][row][1];
      }
    }

    // Rescale O by alpha(row), then accumulate P * V^T (16 WMMAs).
    float al[8];
#pragma unroll
    for (int r8 = 0; r8 < 8; ++r8) al[r8] = sAl[wave][r8 + 8 * half];
#pragma unroll
    for (int dt = 0; dt < 8; ++dt)
#pragma unroll
      for (int r8 = 0; r8 < 8; ++r8) O[dt][r8] *= al[r8];

#pragma unroll
    for (int ks2 = 0; ks2 < 2; ++ks2) {
      const bf16x16 pf = load_frag_a(&sP[wave][ks2 * 32], FLP, lane);
#pragma unroll
      for (int dt = 0; dt < 8; ++dt) {
        const bf16x16 vf = load_frag_b(cV + dt * 16 * FLV + ks2 * 32, FLV, lane);
        O[dt] = wmma_bf16(pf, vf, O[dt]);
      }
    }
    __syncthreads();          // protect buf[cur^1] before next prefetch refills it
  }

  // Finalize: O /= l(row); store bf16 to [B,S,H*D].
  float li[8];
#pragma unroll
  for (int r8 = 0; r8 < 8; ++r8) li[r8] = 1.0f / sL[wave][r8 + 8 * half];
  bf16* obase = out + ((size_t)(b * SS + qi * 64 + wave * 16) * (HH * DD)) + h * DD;
#pragma unroll
  for (int dt = 0; dt < 8; ++dt)
#pragma unroll
    for (int r8 = 0; r8 < 8; ++r8) {
      const int row = r8 + 8 * half;
      obase[(size_t)row * (HH * DD) + dt * 16 + cin] = to_bf16(O[dt][r8] * li[r8]);
    }
}

// ---------------------------------------------------------------------------
// LayerNorm (f32 in -> bf16 out). One wave32 per row of 128 elements.
// ---------------------------------------------------------------------------
__global__ __launch_bounds__(128) void layernorm_bf16(
    const float* __restrict__ x, const float* __restrict__ g,
    const float* __restrict__ bta, bf16* __restrict__ out)
{
  const int row = blockIdx.x * 4 + (threadIdx.x >> 5);
  const int lane = threadIdx.x & 31;
  const float* p = x + (size_t)row * EE;
  float v[4], s = 0.0f;
#pragma unroll
  for (int j = 0; j < 4; ++j) { v[j] = p[lane + 32 * j]; s += v[j]; }
#pragma unroll
  for (int off = 16; off > 0; off >>= 1) s += __shfl_xor(s, off, 32);
  const float mu = s * (1.0f / EE);
  float var = 0.0f;
#pragma unroll
  for (int j = 0; j < 4; ++j) { const float d = v[j] - mu; var += d * d; }
#pragma unroll
  for (int off = 16; off > 0; off >>= 1) var += __shfl_xor(var, off, 32);
  const float rinv = rsqrtf(var * (1.0f / EE) + 1e-5f);
  bf16* o = out + (size_t)row * EE;
#pragma unroll
  for (int j = 0; j < 4; ++j) {
    const int c = lane + 32 * j;
    o[c] = to_bf16((v[j] - mu) * rinv * g[c] + bta[c]);
  }
}

// RoPE (f32 [B,S,H,D] -> bf16, interleaved even/odd pairs). 64 threads = 64 pairs.
__global__ void rope_bf16(const float* __restrict__ src, bf16* __restrict__ dst)
{
  const int idx = blockIdx.x;              // (b*S + s)*H + h
  const int s = (idx / HH) % SS;
  const int i = threadIdx.x;               // pair index 0..63
  const float* p = src + (size_t)idx * DD;
  const float x1 = p[2 * i], x2 = p[2 * i + 1];
  const float inv = __expf(-(float)i * 0.14391156855f);  // ln(10000)/64
  float sn, cs;
  __sincosf((float)s * inv, &sn, &cs);
  dst[(size_t)idx * DD + 2 * i]     = to_bf16(x1 * cs - x2 * sn);
  dst[(size_t)idx * DD + 2 * i + 1] = to_bf16(x1 * sn + x2 * cs);
}

// Weight convert + transpose: f32 [K][N] -> bf16 [N][K]
__global__ void wtrans(const float* __restrict__ w, bf16* __restrict__ wt, int K, int N)
{
  const size_t i = (size_t)blockIdx.x * 256 + threadIdx.x;
  if (i >= (size_t)K * N) return;
  const int n = (int)(i % N);
  const int k = (int)(i / N);
  wt[(size_t)n * K + k] = to_bf16(w[i]);
}

// V transpose: bf16 [B,S,H,D] -> bf16 [B,H,D,S]
__global__ void vtrans(const bf16* __restrict__ v, bf16* __restrict__ vtp)
{
  const size_t i = (size_t)blockIdx.x * 256 + threadIdx.x;   // exact grid
  const int d = (int)(i % DD);
  size_t t = i / DD;
  const int h = (int)(t % HH); t /= HH;
  const int s = (int)(t % SS);
  const int b = (int)(t / SS);
  vtp[(((size_t)b * HH + h) * DD + d) * SS + s] = v[i];
}

// ---------------------------------------------------------------------------
extern "C" void kernel_launch(void* const* d_in, const int* in_sizes, int n_in,
                              void* d_out, int out_size, void* d_ws, size_t ws_size,
                              hipStream_t stream)
{
  const float* x      = (const float*)d_in[0];
  const float* W_q    = (const float*)d_in[1];
  const float* W_k    = (const float*)d_in[2];
  const float* W_v    = (const float*)d_in[3];
  const float* W_o    = (const float*)d_in[4];
  const float* ln1_g  = (const float*)d_in[5];
  const float* ln1_b  = (const float*)d_in[6];
  const float* fc1_w  = (const float*)d_in[7];
  const float* fc1_b  = (const float*)d_in[8];
  const float* fc2_w  = (const float*)d_in[9];
  const float* fc2_b  = (const float*)d_in[10];
  const float* ln2_g  = (const float*)d_in[11];
  const float* ln2_b  = (const float*)d_in[12];

  char* ws = (char*)d_ws;
  size_t off = 0;
  auto alloc = [&](size_t bytes) -> void* {
    void* p = ws + off;
    off = (off + bytes + 255) & ~(size_t)255;
    return p;
  };
  float* qkf  = (float*)alloc((size_t)MM * 2048 * 4);  // pre-RoPE q/k (reused)
  bf16*  qbf  = (bf16*) alloc((size_t)MM * 2048 * 2);
  bf16*  kbf2 = (bf16*) alloc((size_t)MM * 2048 * 2);
  bf16*  vbf  = (bf16*) alloc((size_t)MM * 2048 * 2);
  bf16*  vT   = (bf16*) alloc((size_t)MM * 2048 * 2);
  bf16*  attn = (bf16*) alloc((size_t)MM * 2048 * 2);
  float* x2   = (float*)alloc((size_t)MM * EE * 4);
  bf16*  hln  = (bf16*) alloc((size_t)MM * EE * 2);
  bf16*  h2ln = (bf16*) alloc((size_t)MM * EE * 2);
  bf16*  gbuf = (bf16*) alloc((size_t)MM * DHH * 2);
  bf16*  Wqt  = (bf16*) alloc((size_t)2048 * 128 * 2);
  bf16*  Wkt  = (bf16*) alloc((size_t)2048 * 128 * 2);
  bf16*  Wvt  = (bf16*) alloc((size_t)2048 * 128 * 2);
  bf16*  Wot  = (bf16*) alloc((size_t)128 * 2048 * 2);
  bf16*  f1t  = (bf16*) alloc((size_t)512 * 128 * 2);
  bf16*  f2t  = (bf16*) alloc((size_t)128 * 512 * 2);
  (void)in_sizes; (void)n_in; (void)out_size; (void)ws_size;

  // Weight conversion (f32 [K][N] -> bf16 [N][K])
  wtrans<<<dim3((128 * 2048) / 256), 256, 0, stream>>>(W_q, Wqt, 128, 2048);
  wtrans<<<dim3((128 * 2048) / 256), 256, 0, stream>>>(W_k, Wkt, 128, 2048);
  wtrans<<<dim3((128 * 2048) / 256), 256, 0, stream>>>(W_v, Wvt, 128, 2048);
  wtrans<<<dim3((2048 * 128) / 256), 256, 0, stream>>>(W_o, Wot, 2048, 128);
  wtrans<<<dim3((128 * 512) / 256),  256, 0, stream>>>(fc1_w, f1t, 128, 512);
  wtrans<<<dim3((512 * 128) / 256),  256, 0, stream>>>(fc2_w, f2t, 512, 128);

  // LN1
  layernorm_bf16<<<dim3(MM / 4), 128, 0, stream>>>(x, ln1_g, ln1_b, hln);

  // Q = hln @ Wq -> RoPE -> bf16
  gemm_wmma<EP_F32><<<dim3(2048 / 64, MM / 64), 128, 0, stream>>>(
      hln, Wqt, MM, 2048, 128, nullptr, nullptr, qkf, nullptr);
  rope_bf16<<<dim3(MM * HH), 64, 0, stream>>>(qkf, qbf);
  // K
  gemm_wmma<EP_F32><<<dim3(2048 / 64, MM / 64), 128, 0, stream>>>(
      hln, Wkt, MM, 2048, 128, nullptr, nullptr, qkf, nullptr);
  rope_bf16<<<dim3(MM * HH), 64, 0, stream>>>(qkf, kbf2);
  // V (bf16 direct) + transpose to [B,H,D,S]
  gemm_wmma<EP_BF16><<<dim3(2048 / 64, MM / 64), 128, 0, stream>>>(
      hln, Wvt, MM, 2048, 128, nullptr, nullptr, nullptr, vbf);
  vtrans<<<dim3((MM * 2048) / 256), 256, 0, stream>>>(vbf, vT);

  // Causal flash attention
  flash_attn<<<dim3(SS / 64, BB * HH), 128, 0, stream>>>(qbf, kbf2, vT, attn);

  // x2 = x + attn @ W_o
  gemm_wmma<EP_RES_F32><<<dim3(128 / 64, MM / 64), 128, 0, stream>>>(
      attn, Wot, MM, 128, 2048, nullptr, x, x2, nullptr);

  // LN2 + MLP with fused bias/GELU/residual
  layernorm_bf16<<<dim3(MM / 4), 128, 0, stream>>>(x2, ln2_g, ln2_b, h2ln);
  gemm_wmma<EP_BIAS_GELU_BF16><<<dim3(512 / 64, MM / 64), 128, 0, stream>>>(
      h2ln, f1t, MM, 512, 128, fc1_b, nullptr, nullptr, gbuf);
  gemm_wmma<EP_BIAS_RES_F32><<<dim3(128 / 64, MM / 64), 128, 0, stream>>>(
      gbuf, f2t, MM, 128, 512, fc2_b, x2, (float*)d_out, nullptr);
}
